// AddingLRU_86930138071300
// MI455X (gfx1250) — matrix operand order repaired and checked by reading, not compiled, not run on hardware
//
#include <hip/hip_runtime.h>
#include <hip/hip_bf16.h>
#include <math.h>

typedef __attribute__((ext_vector_type(16))) __bf16 v16bf;
typedef __attribute__((ext_vector_type(8)))  float  v8f;

#define B_   16
#define T_   4096
#define IN_  2
#define H_   256
#define N_   256
#define CK   64          // chunk length (time steps per chunk)
#define NC   (T_/CK)     // 64 chunks

// ---- workspace layout (float offsets for the small-param region) ----
#define OFF_LAM_RE   0
#define OFF_LAM_IM   256
#define OFF_LAMK_RE  512
#define OFF_LAMK_IM  768
#define OFF_BM0RE    1024
#define OFF_BM0IM    1280
#define OFF_BM1RE    1536
#define OFF_BM1IM    1792
#define OFF_CCRE     2048
#define OFF_CCIM     2304
#define OFF_DD       2560   // 2 floats
// byte offsets for the bf16 power tables and chunk contributions
#define VRE_BYTE         16384
#define VIM_BYTE         (VRE_BYTE + N_*CK*2)
#define CONTRIB_RE_BYTE  (VIM_BYTE + N_*CK*2)              // f32 [B_][NC][N_]
#define CONTRIB_IM_BYTE  (CONTRIB_RE_BYTE + B_*NC*N_*4)    // f32 [B_][NC][N_]
// total ws use: ~16KB + 64KB + 2MB

// -------------------------------------------------------------------
// Kernel 1: parameter setup (1 block, 256 threads; thread = state n)
// lam = exp(-exp(nu_log) + i*exp(theta_log))  (exact closed form for powers)
// -------------------------------------------------------------------
__global__ void __launch_bounds__(256)
lru_setup(const float* nu_log, const float* theta_log,
          const float* B_re, const float* B_im,
          const float* C_re, const float* C_im,
          const float* D, const float* Wh, float* ws) {
    int n = threadIdx.x;
    float nu = __expf(nu_log[n]);
    float th = __expf(theta_log[n]);
    float mag = __expf(-nu);
    ws[OFF_LAM_RE + n] = mag * __cosf(th);
    ws[OFF_LAM_IM + n] = mag * __sinf(th);
    // lam^CK in closed form (no error accumulation)
    float magK = __expf(-(float)CK * nu);
    float thK  = (float)CK * th;
    ws[OFF_LAMK_RE + n] = magK * __cosf(thK);
    ws[OFF_LAMK_IM + n] = magK * __sinf(thK);
    // gamma-scaled input matrix Bm[n][i] (complex), IN_=2
    float gamma = sqrtf(fmaxf(0.0f, 1.0f - __expf(-2.0f * nu)));
    ws[OFF_BM0RE + n] = B_re[n * IN_ + 0] * gamma;
    ws[OFF_BM0IM + n] = B_im[n * IN_ + 0] * gamma;
    ws[OFF_BM1RE + n] = B_re[n * IN_ + 1] * gamma;
    ws[OFF_BM1IM + n] = B_im[n * IN_ + 1] * gamma;
    // power table V[n][j] = lam^(CK-1-j), stored bf16 for WMMA A-operand
    __bf16* Vre = (__bf16*)((char*)ws + VRE_BYTE);
    __bf16* Vim = (__bf16*)((char*)ws + VIM_BYTE);
    for (int j = 0; j < CK; ++j) {
        float p = (float)(CK - 1 - j);
        float m = __expf(-p * nu);
        Vre[n * CK + j] = (__bf16)(m * __cosf(p * th));
        Vim[n * CK + j] = (__bf16)(m * __sinf(p * th));
    }
    // fold output head: CC[n] = sum_h Wh[h] * (C_re + i C_im)[h][n]
    float ccre = 0.0f, ccim = 0.0f;
    for (int h = 0; h < H_; ++h) {
        float w = Wh[h];
        ccre += w * C_re[h * N_ + n];
        ccim += w * C_im[h * N_ + n];
    }
    ws[OFF_CCRE + n] = ccre;
    ws[OFF_CCIM + n] = ccim;
    if (n < IN_) {
        float dd = 0.0f;
        for (int h = 0; h < H_; ++h) dd += Wh[h] * D[h * IN_ + n];
        ws[OFF_DD + n] = dd;
    }
}

// -------------------------------------------------------------------
// WMMA operand gathers (wave32; contiguous k order used by this code, same for A and B)
// A 16x32: lane L (m = L&15): elem e -> K = e + 16*(L>=16) (+32*kk)
// B 32x16: lane L (ncol = L&15): elem e -> K = e + 16*(L>=16) (+32*kk)
// -------------------------------------------------------------------
__device__ inline v16bf load_V_tile(const __bf16* V, int n0, int lane, int kk) {
    int lanehi = (lane >> 4) & 1;
    int lmod   = lane & 15;
    const __bf16* row = V + (n0 + lmod) * CK + 32 * kk + 16 * lanehi;
    v16bf a;
#pragma unroll
    for (int e = 0; e < 16; ++e) {
        int k = e;
        a[e] = row[k];
    }
    return a;
}

__device__ inline v16bf load_X_tile(const float* x, int b, int c0, int lane,
                                    int i, int kk) {
    int lanehi = (lane >> 4) & 1;
    int lmod   = lane & 15;
    int c      = c0 + lmod;
    int tbase  = c * CK + 32 * kk + 16 * lanehi;
    const float* px = x + ((size_t)b * T_ + tbase) * IN_ + i;
    v16bf v;
#pragma unroll
    for (int e = 0; e < 16; ++e) v[e] = (__bf16)px[e * IN_];
    return v;
}

// -------------------------------------------------------------------
// Kernel 2: chunk contributions via WMMA.
// One wave per 16(n) x 16(chunk) output tile; 8x v_wmma_f32_16x16x32_bf16.
// contrib[b,c,n] = sum_i Bm[n][i] * sum_j lam^(CK-1-j) * x[b, c*CK+j, i]
// -------------------------------------------------------------------
__global__ void __launch_bounds__(32)
lru_chunk_wmma(const float* __restrict__ x, float* __restrict__ ws) {
    int lane = threadIdx.x;
    int ct = blockIdx.x;   // chunk tile 0..3
    int nt = blockIdx.y;   // state tile 0..15
    int b  = blockIdx.z;   // batch 0..15
    const __bf16* Vre = (const __bf16*)((const char*)ws + VRE_BYTE);
    const __bf16* Vim = (const __bf16*)((const char*)ws + VIM_BYTE);

    v16bf ar0 = load_V_tile(Vre, nt * 16, lane, 0);
    v16bf ar1 = load_V_tile(Vre, nt * 16, lane, 1);
    v16bf ai0 = load_V_tile(Vim, nt * 16, lane, 0);
    v16bf ai1 = load_V_tile(Vim, nt * 16, lane, 1);
    v16bf b00 = load_X_tile(x, b, ct * 16, lane, 0, 0);
    v16bf b01 = load_X_tile(x, b, ct * 16, lane, 0, 1);
    v16bf b10 = load_X_tile(x, b, ct * 16, lane, 1, 0);
    v16bf b11 = load_X_tile(x, b, ct * 16, lane, 1, 1);

    v8f p0r = {}, p0i = {}, p1r = {}, p1i = {};
    p0r = __builtin_amdgcn_wmma_f32_16x16x32_bf16(false, ar0, false, b00, (short)0, p0r, false, false);
    p0r = __builtin_amdgcn_wmma_f32_16x16x32_bf16(false, ar1, false, b01, (short)0, p0r, false, false);
    p0i = __builtin_amdgcn_wmma_f32_16x16x32_bf16(false, ai0, false, b00, (short)0, p0i, false, false);
    p0i = __builtin_amdgcn_wmma_f32_16x16x32_bf16(false, ai1, false, b01, (short)0, p0i, false, false);
    p1r = __builtin_amdgcn_wmma_f32_16x16x32_bf16(false, ar0, false, b10, (short)0, p1r, false, false);
    p1r = __builtin_amdgcn_wmma_f32_16x16x32_bf16(false, ar1, false, b11, (short)0, p1r, false, false);
    p1i = __builtin_amdgcn_wmma_f32_16x16x32_bf16(false, ai0, false, b10, (short)0, p1i, false, false);
    p1i = __builtin_amdgcn_wmma_f32_16x16x32_bf16(false, ai1, false, b11, (short)0, p1i, false, false);

    // C/D layout: VGPR r, lane<16 -> row r, lane>=16 -> row r+8; col = lane&15
    int lanehi = (lane >> 4) & 1;
    int lmod   = lane & 15;
    int c      = ct * 16 + lmod;
    float* cr = (float*)((char*)ws + CONTRIB_RE_BYTE);
    float* ci = (float*)((char*)ws + CONTRIB_IM_BYTE);
#pragma unroll
    for (int r = 0; r < 8; ++r) {
        int n = nt * 16 + r + 8 * lanehi;
        float b0r = ws[OFF_BM0RE + n], b0i = ws[OFF_BM0IM + n];
        float b1r = ws[OFF_BM1RE + n], b1i = ws[OFF_BM1IM + n];
        float cre = b0r * p0r[r] - b0i * p0i[r] + b1r * p1r[r] - b1i * p1i[r];
        float cim = b0i * p0r[r] + b0r * p0i[r] + b1i * p1r[r] + b1r * p1i[r];
        size_t idx = ((size_t)b * NC + c) * N_ + n;
        cr[idx] = cre;
        ci[idx] = cim;
    }
}

// -------------------------------------------------------------------
// Kernel 3: per-batch chunk combine (depth <= 64) + remainder (< 64 steps)
// + folded output head reduction. Block = 256 threads (thread = n).
// -------------------------------------------------------------------
__global__ void __launch_bounds__(256)
lru_combine(const float* __restrict__ x, const int* __restrict__ lengths,
            const float* __restrict__ bh, const float* __restrict__ ws,
            float* __restrict__ out) {
    int b = blockIdx.x;
    int n = threadIdx.x;
    int L = lengths[b];          // in [1, T_]
    int Cfull = L >> 6;          // full chunks fully inside [0, L)
    float lKre = ws[OFF_LAMK_RE + n], lKim = ws[OFF_LAMK_IM + n];
    const float* cr = (const float*)((const char*)ws + CONTRIB_RE_BYTE);
    const float* ci = (const float*)((const char*)ws + CONTRIB_IM_BYTE);
    float sre = 0.0f, sim = 0.0f;
    for (int c = 0; c < Cfull; ++c) {
        size_t idx = ((size_t)b * NC + c) * N_ + n;
        float tre = lKre * sre - lKim * sim + cr[idx];
        float tim = lKre * sim + lKim * sre + ci[idx];
        sre = tre; sim = tim;
    }
    // remainder steps: direct recurrence s = lam*s + Bm @ x_t
    float lre = ws[OFF_LAM_RE + n], lim = ws[OFF_LAM_IM + n];
    float b0r = ws[OFF_BM0RE + n], b0i = ws[OFF_BM0IM + n];
    float b1r = ws[OFF_BM1RE + n], b1i = ws[OFF_BM1IM + n];
    for (int t = Cfull << 6; t < L; ++t) {
        float x0 = x[((size_t)b * T_ + t) * IN_ + 0];
        float x1 = x[((size_t)b * T_ + t) * IN_ + 1];
        float bre = x0 * b0r + x1 * b1r;
        float bim = x0 * b0i + x1 * b1i;
        float tre = lre * sre - lim * sim + bre;
        float tim = lre * sim + lim * sre + bim;
        sre = tre; sim = tim;
    }
    // out[b] = Re(sum_n s_n * CC[n]) + x[b,L-1,:].DD + bh
    float val = sre * ws[OFF_CCRE + n] - sim * ws[OFF_CCIM + n];
    __shared__ float red[256];
    red[n] = val;
    __syncthreads();
    for (int s = 128; s > 0; s >>= 1) {
        if (n < s) red[n] += red[n + s];
        __syncthreads();
    }
    if (n == 0) {
        float x0 = x[((size_t)b * T_ + (L - 1)) * IN_ + 0];
        float x1 = x[((size_t)b * T_ + (L - 1)) * IN_ + 1];
        out[b] = red[0] + x0 * ws[OFF_DD + 0] + x1 * ws[OFF_DD + 1] + bh[0];
    }
}

extern "C" void kernel_launch(void* const* d_in, const int* in_sizes, int n_in,
                              void* d_out, int out_size, void* d_ws, size_t ws_size,
                              hipStream_t stream) {
    (void)in_sizes; (void)n_in; (void)out_size; (void)ws_size;
    const float* x         = (const float*)d_in[0];
    const int*   lengths   = (const int*)d_in[1];
    const float* nu_log    = (const float*)d_in[2];
    const float* theta_log = (const float*)d_in[3];
    const float* B_re      = (const float*)d_in[4];
    const float* B_im      = (const float*)d_in[5];
    const float* C_re      = (const float*)d_in[6];
    const float* C_im      = (const float*)d_in[7];
    const float* D         = (const float*)d_in[8];
    const float* Wh        = (const float*)d_in[9];
    const float* bh        = (const float*)d_in[10];
    float* ws  = (float*)d_ws;
    float* out = (float*)d_out;

    hipLaunchKernelGGL(lru_setup, dim3(1), dim3(256), 0, stream,
                       nu_log, theta_log, B_re, B_im, C_re, C_im, D, Wh, ws);
    hipLaunchKernelGGL(lru_chunk_wmma, dim3(NC / 16, N_ / 16, B_), dim3(32), 0, stream,
                       x, ws);
    hipLaunchKernelGGL(lru_combine, dim3(B_), dim3(256), 0, stream,
                       x, lengths, bh, ws, out);
}
